// ECG_RNN_Classifier_8151847928514
// MI455X (gfx1250) — matrix operand order repaired and checked by reading, not compiled, not run on hardware
//
#include <hip/hip_runtime.h>
#include <stdint.h>

#define BB 1024
#define TT 187
#define HH 256
#define CC 5

typedef __attribute__((ext_vector_type(16))) __bf16 bf16x16;
typedef __attribute__((ext_vector_type(8)))  float  f32x8;

struct Bytes32  { uint4 lo, hi; };          // 32 bytes
struct Shorts16 { unsigned short s[16]; };  // 32 bytes

// native f32 -> bf16 conversion (v_cvt on gfx1250), RNE
static __device__ __forceinline__ unsigned short f2bf_bits(float f) {
  __bf16 b = (__bf16)f;
  return __builtin_bit_cast(unsigned short, b);
}

// hardware tanh (gfx1250 V_TANH_F32)
#if defined(__has_builtin)
#if __has_builtin(__builtin_amdgcn_tanhf)
#define FAST_TANH(x) __builtin_amdgcn_tanhf(x)
#elif __has_builtin(__builtin_amdgcn_tanh_f32)
#define FAST_TANH(x) __builtin_amdgcn_tanh_f32(x)
#endif
#endif
#ifndef FAST_TANH
static __device__ __forceinline__ float fast_tanh_asm(float x) {
  float r;
  asm volatile("v_tanh_f32 %0, %1\n\tv_nop" : "=v"(r) : "v"(x));
  return r;
}
#define FAST_TANH(x) fast_tanh_asm(x)
#endif

// ---------------------------------------------------------------------------
// Layer 0: h_t = tanh(x[b,t]*W_ih0 + b + h_{t-1} @ W_hh0^T), write hs0 (bf16)
// grid = 64 (batch tiles of 16 rows), block = 512 (16 waves, one 16x16 N-tile
// each). W_hh0 bf16 B-fragments live in VGPRs for the whole T loop.
// Double-buffered h in LDS -> ONE barrier per step; hs0 writeback via
// global_store_async_from_lds_b128 draining behind the next step's compute.
// ---------------------------------------------------------------------------
__global__ __launch_bounds__(512) void rnn_layer0(
    const float* __restrict__ x,       // [B,T]
    const float* __restrict__ W_ih0,   // [H,1]
    const float* __restrict__ W_hh0,   // [H,H] (row = output neuron)
    const float* __restrict__ b_ih0,
    const float* __restrict__ b_hh0,
    unsigned short* __restrict__ hs0)  // [T,B,H] bf16 bits
{
  __shared__ unsigned short hbuf[2][16 * HH];  // 2 x 8 KB ping-pong h state
  __shared__ float xtile[16 * TT];             // ~12 KB staged input rows

  const int tid   = threadIdx.x;
  const int wave  = tid >> 5;
  const int lane  = tid & 31;
  const int lrow  = lane & 15;
  const int lhalf = lane >> 4;
  const int b0    = blockIdx.x * 16;
  const int n     = wave * 16 + lrow;

  for (int i = tid; i < 16 * TT; i += 512)
    xtile[i] = x[(size_t)(b0 + i / TT) * TT + (i % TT)];
  for (int i = tid; i < 16 * HH; i += 512) hbuf[0][i] = 0;   // h_{-1} = 0

  // B-fragments of W_hh0 (bf16): per lane two runs of 8 K-values at
  // row n (ISA 16-bit B layout, matching the A table order).
  bf16x16 wf[8];
  #pragma unroll
  for (int kt = 0; kt < 8; ++kt) {
    const float* wp = W_hh0 + (size_t)n * HH + kt * 32 + lhalf * 8;
    Shorts16 tmp;
    #pragma unroll
    for (int j = 0; j < 16; ++j) tmp.s[j] = f2bf_bits(wp[(j & 7) + (j >> 3) * 16]);
    wf[kt] = __builtin_bit_cast(bf16x16, tmp);
  }

  const float bias = b_ih0[n] + b_hh0[n];
  const float wih  = W_ih0[n];

  // async writeback: thread owns one 16B chunk of the h tile
  const uint32_t lds_h[2] = {
    (uint32_t)(uintptr_t)(&hbuf[0][0]) + tid * 16,
    (uint32_t)(uintptr_t)(&hbuf[1][0]) + tid * 16 };
  uint64_t gaddr = (uint64_t)(uintptr_t)hs0 +
                   ((size_t)(b0 + wave) * HH) * 2 + lane * 16;
  const uint64_t gstride = (uint64_t)BB * HH * 2;

  __syncthreads();

  for (int t = 0; t < TT; ++t) {
    const int rp = t & 1;                       // read hbuf[rp], write hbuf[rp^1]
    if (t > 0) {                                // async store h_{t-1} (same buf as
      asm volatile("global_store_async_from_lds_b128 %0, %1, off"
                   :: "v"(gaddr), "v"(rp ? lds_h[1] : lds_h[0]) : "memory");
      gaddr += gstride;                         //  A-reads: reader/reader, safe)
    }

    // A-fragments of h_{t-1} (16-bit A 16x32 layout: two 16B runs per k-tile)
    bf16x16 af[8];
    const char* abase = (const char*)&hbuf[rp][0] + lrow * (HH * 2);
    #pragma unroll
    for (int kt = 0; kt < 8; ++kt) {
      Bytes32 c;
      c.lo = *(const uint4*)(abase + kt * 64 + lhalf * 16);
      c.hi = *(const uint4*)(abase + kt * 64 + 32 + lhalf * 16);
      af[kt] = __builtin_bit_cast(bf16x16, c);
    }

    // two independent WMMA chains for XDL pipelining
    f32x8 accA, accB;
    #pragma unroll
    for (int v = 0; v < 8; ++v) {
      const int m = v + 8 * lhalf;             // C/D layout row
      accA[v] = xtile[m * TT + t] * wih + bias;
      accB[v] = 0.0f;
    }
    #pragma unroll
    for (int kt = 0; kt < 4; ++kt) {
      accA = __builtin_amdgcn_wmma_f32_16x16x32_bf16(false, af[kt], false, wf[kt],
                                                     (short)0, accA, false, false);
      accB = __builtin_amdgcn_wmma_f32_16x16x32_bf16(false, af[kt + 4], false, wf[kt + 4],
                                                     (short)0, accB, false, false);
    }

    unsigned short hb[8];
    #pragma unroll
    for (int v = 0; v < 8; ++v) hb[v] = f2bf_bits(FAST_TANH(accA[v] + accB[v]));

    #pragma unroll
    for (int v = 0; v < 8; ++v) hbuf[rp ^ 1][(v + 8 * lhalf) * HH + n] = hb[v];

    // own async store (issued this step) has drained behind the compute;
    // barrier then publishes h_t and all async completions to every wave.
    asm volatile("s_wait_asynccnt 0" ::: "memory");
    __syncthreads();
  }
  // final h_{T-1} lives in hbuf[TT & 1]; S_ENDPGM's implicit wait-idle flushes.
  asm volatile("global_store_async_from_lds_b128 %0, %1, off"
               :: "v"(gaddr), "v"(lds_h[TT & 1]) : "memory");
}

// ---------------------------------------------------------------------------
// Layer 1: h_t = tanh(hs0_t @ W_ih1^T + b + h_{t-1} @ W_hh1^T)
// hs0 tiles double-buffered via global_load_async_to_lds_b128; both weight
// matrices in VGPRs; double-buffered h -> ONE barrier per step; the two
// matmuls run as independent WMMA chains.
// ---------------------------------------------------------------------------
__global__ __launch_bounds__(512) void rnn_layer1(
    const unsigned short* __restrict__ hs0,  // [T,B,H] bf16 bits
    const float* __restrict__ W_ih1,
    const float* __restrict__ W_hh1,
    const float* __restrict__ b_ih1,
    const float* __restrict__ b_hh1,
    float* __restrict__ hlast)               // [B,H] f32
{
  __shared__ unsigned short hbuf[2][16 * HH];    // 2 x 8 KB recurrent state
  __shared__ unsigned short xbuf[2][16 * HH];    // 2 x 8 KB hs0_t staging

  const int tid   = threadIdx.x;
  const int wave  = tid >> 5;
  const int lane  = tid & 31;
  const int lrow  = lane & 15;
  const int lhalf = lane >> 4;
  const int b0    = blockIdx.x * 16;
  const int n     = wave * 16 + lrow;

  for (int i = tid; i < 16 * HH; i += 512) hbuf[0][i] = 0;   // h_{-1} = 0

  bf16x16 wih[8], whh[8];
  #pragma unroll
  for (int kt = 0; kt < 8; ++kt) {
    const float* p0 = W_ih1 + (size_t)n * HH + kt * 32 + lhalf * 8;
    const float* p1 = W_hh1 + (size_t)n * HH + kt * 32 + lhalf * 8;
    Shorts16 t0, t1;
    #pragma unroll
    for (int j = 0; j < 16; ++j) { const int o = (j & 7) + (j >> 3) * 16; t0.s[j] = f2bf_bits(p0[o]); t1.s[j] = f2bf_bits(p1[o]); }
    wih[kt] = __builtin_bit_cast(bf16x16, t0);
    whh[kt] = __builtin_bit_cast(bf16x16, t1);
  }

  const float bias = b_ih1[n] + b_hh1[n];

  const uint32_t lds_x[2] = {
    (uint32_t)(uintptr_t)(&xbuf[0][0]) + tid * 16,
    (uint32_t)(uintptr_t)(&xbuf[1][0]) + tid * 16 };
  uint64_t gaddr = (uint64_t)(uintptr_t)hs0 +
                   ((size_t)(b0 + wave) * HH) * 2 + lane * 16;
  const uint64_t gstride = (uint64_t)BB * HH * 2;

  // prefetch hs0[t=0] tile into xbuf[0]
  asm volatile("global_load_async_to_lds_b128 %0, %1, off"
               :: "v"(lds_x[0]), "v"(gaddr) : "memory");
  gaddr += gstride;
  asm volatile("s_wait_asynccnt 0" ::: "memory");
  __syncthreads();   // hbuf[0] zeros + xbuf[0] visible

  for (int t = 0; t < TT; ++t) {
    const int rp = t & 1;                      // read xbuf[rp]/hbuf[rp], write hbuf[rp^1]
    if (t + 1 < TT) {                          // prefetch next tile behind compute
      asm volatile("global_load_async_to_lds_b128 %0, %1, off"
                   :: "v"(rp ? lds_x[0] : lds_x[1]), "v"(gaddr) : "memory");
      gaddr += gstride;
    }

    f32x8 accX, accH;
    #pragma unroll
    for (int v = 0; v < 8; ++v) { accX[v] = bias; accH[v] = 0.0f; }

    const char* xb = (const char*)&xbuf[rp][0] + lrow * (HH * 2);
    const char* ab = (const char*)&hbuf[rp][0] + lrow * (HH * 2);
    #pragma unroll
    for (int kt = 0; kt < 8; ++kt) {
      Bytes32 cx, ch;
      cx.lo = *(const uint4*)(xb + kt * 64 + lhalf * 16);
      cx.hi = *(const uint4*)(xb + kt * 64 + 32 + lhalf * 16);
      ch.lo = *(const uint4*)(ab + kt * 64 + lhalf * 16);
      ch.hi = *(const uint4*)(ab + kt * 64 + 32 + lhalf * 16);
      accX = __builtin_amdgcn_wmma_f32_16x16x32_bf16(false, __builtin_bit_cast(bf16x16, cx),
                                                     false, wih[kt], (short)0, accX, false, false);
      accH = __builtin_amdgcn_wmma_f32_16x16x32_bf16(false, __builtin_bit_cast(bf16x16, ch),
                                                     false, whh[kt], (short)0, accH, false, false);
    }

    float hv[8];
    #pragma unroll
    for (int v = 0; v < 8; ++v) hv[v] = FAST_TANH(accX[v] + accH[v]);

    #pragma unroll
    for (int v = 0; v < 8; ++v) hbuf[rp ^ 1][(v + 8 * lhalf) * HH + n] = f2bf_bits(hv[v]);
    if (t == TT - 1) {
      #pragma unroll
      for (int v = 0; v < 8; ++v)
        hlast[(size_t)(b0 + v + 8 * lhalf) * HH + n] = hv[v];
    }

    // own prefetch for t+1 has landed; barrier publishes h_t + next tile.
    asm volatile("s_wait_asynccnt 0" ::: "memory");
    __syncthreads();
  }
}

// ---------------------------------------------------------------------------
// Classifier: out[b,c] = hlast[b,:] . W_fc[c,:] + b_fc[c]   (trivial, 1.3 MFLOP)
// ---------------------------------------------------------------------------
__global__ void fc_kernel(const float* __restrict__ hlast,
                          const float* __restrict__ W_fc,
                          const float* __restrict__ b_fc,
                          float* __restrict__ out)
{
  int gid = blockIdx.x * blockDim.x + threadIdx.x;
  if (gid >= BB * CC) return;
  int b = gid / CC, c = gid % CC;
  const float* hp = hlast + (size_t)b * HH;
  const float* wp = W_fc + (size_t)c * HH;
  float acc = b_fc[c];
  #pragma unroll 8
  for (int k = 0; k < HH; ++k) acc += hp[k] * wp[k];
  out[gid] = acc;
}

extern "C" void kernel_launch(void* const* d_in, const int* in_sizes, int n_in,
                              void* d_out, int out_size, void* d_ws, size_t ws_size,
                              hipStream_t stream) {
  const float* x     = (const float*)d_in[0];
  const float* W_ih0 = (const float*)d_in[1];
  const float* W_hh0 = (const float*)d_in[2];
  const float* b_ih0 = (const float*)d_in[3];
  const float* b_hh0 = (const float*)d_in[4];
  const float* W_ih1 = (const float*)d_in[5];
  const float* W_hh1 = (const float*)d_in[6];
  const float* b_ih1 = (const float*)d_in[7];
  const float* b_hh1 = (const float*)d_in[8];
  const float* W_fc  = (const float*)d_in[9];
  const float* b_fc  = (const float*)d_in[10];

  unsigned short* hs0 = (unsigned short*)d_ws;                        // 98 MB bf16
  float* hlast = (float*)((char*)d_ws + (size_t)TT * BB * HH * 2);    // 1 MB f32

  rnn_layer0<<<BB / 16, 512, 0, stream>>>(x, W_ih0, W_hh0, b_ih0, b_hh0, hs0);
  rnn_layer1<<<BB / 16, 512, 0, stream>>>(hs0, W_ih1, W_hh1, b_ih1, b_hh1, hlast);
  fc_kernel<<<(BB * CC + 255) / 256, 256, 0, stream>>>(hlast, W_fc, b_fc, (float*)d_out);
}
